// OnnxRelPosMultiHeadedAttention_21749714387026
// MI455X (gfx1250) — matrix operand order repaired and checked by reading, not compiled, not run on hardware
//
#include <hip/hip_runtime.h>
#include <stdint.h>

// Problem constants (reference: B=4, T=1024, F=512, H=8, DK=64, P=2*T-1)
#define BB   4
#define TT   1024
#define FF   512
#define HH   8
#define DKK  64
#define PPAD 2048   // P=2047 padded to 2048 (pad row/col = 0)

#define BSTRIDE 72  // LDS B-panel row stride in bf16 (64 data + 8 pad)

typedef __bf16 bf16_t;
typedef bf16_t v16bf __attribute__((ext_vector_type(16)));
typedef bf16_t v8bf  __attribute__((ext_vector_type(8)));
typedef float  v8f   __attribute__((ext_vector_type(8)));

union V16 { v16bf v; v8bf h[2]; };

// ---- WMMA tile loaders (CDNA5 wave32 VGPR layouts, CDNA5 ISA) ----

// 16-bit A 16x32 (MxK): lanes 0-15 hold M=lane, K={0..7,16..23}; lanes 16-31
// hold M=lane-16, K={8..15,24..31}.  Row-major A, K contiguous.
__device__ __forceinline__ v16bf load_a_tile(const bf16_t* base, int lda,
                                             int row0, int k0, int lane) {
  int m  = lane & 15;
  int kh = (lane >> 4) << 3;
  const bf16_t* p = base + (size_t)(row0 + m) * lda + k0 + kh;
  V16 u;
  u.h[0] = *(const v8bf*)(p);
  u.h[1] = *(const v8bf*)(p + 16);
  return u.v;
}

// 16-bit B 32x16 (KxN): lanes hold N=lane&15, K={0..7,16..23} / {8..15,24..31}.
__device__ __forceinline__ v16bf load_b_tile(const bf16_t* base, int ldb,
                                             int k0, int col0, int lane) {
  int n  = lane & 15;
  int kh = (lane >> 4) << 3;
  v16bf r;
#pragma unroll
  for (int e = 0; e < 8; ++e) {
    r[e]     = base[(size_t)(k0 + kh + e) * ldb + col0 + n];
    r[e + 8] = base[(size_t)(k0 + 16 + kh + e) * ldb + col0 + n];
  }
  return r;
}

__device__ __forceinline__ v8f wmma_bf16(v16bf a, v16bf b, v8f c) {
  // 8 args: (neg_a, A, neg_b, B, c_mod, C, reuse_a, reuse_b)
  return __builtin_amdgcn_wmma_f32_16x16x32_bf16(false, a, false, b,
                                                 (short)0, c, false, false);
}

// ---- CDNA5 async global->LDS copy (ASYNCcnt path, CDNA5 ISA) ----

__device__ __forceinline__ void async_b128_to_lds(uint32_t lds_addr,
                                                  const void* gaddr) {
  // GV mode: 64-bit vaddr in VGPR pair, LDS byte address in VDST VGPR.
  asm volatile("global_load_async_to_lds_b128 %0, %1, off"
               :: "v"(lds_addr), "v"(gaddr) : "memory");
}

__device__ __forceinline__ void wait_asynccnt_le1() {
  asm volatile("s_wait_asynccnt 0x1" ::: "memory");
}
__device__ __forceinline__ void wait_asynccnt_0() {
  asm volatile("s_wait_asynccnt 0x0" ::: "memory");
}

__device__ __forceinline__ uint32_t lds_addr_of(const void* p) {
  // LDS aperture: low 32 bits of the flat address select the LDS offset.
  return (uint32_t)(uintptr_t)p;
}

// One k-stage: 2 A tiles (rows row0, row0+16), 4 B tiles from the staged LDS
// panel, 8 WMMAs.  B LDS reads are hoisted into registers first so the
// compiler can batch dscnt waits instead of one wait per WMMA.
__device__ __forceinline__ void stage_compute(
    const bf16_t* __restrict__ A, int lda, int row0, int k, int lane,
    const bf16_t (*bufS)[BSTRIDE], v8f acc[2][4]) {
  v16bf a0 = load_a_tile(A, lda, row0, k, lane);
  v16bf a1 = load_a_tile(A, lda, row0 + 16, k, lane);
  v16bf b[4];
#pragma unroll
  for (int j = 0; j < 4; ++j) {
    int n = lane & 15, kh = (lane >> 4) << 3;
    v16bf t;                        // column n, K={kh..kh+7, 16+kh..16+kh+7}
#pragma unroll
    for (int e = 0; e < 8; ++e) {
      t[e]     = bufS[kh + e][16 * j + n];
      t[e + 8] = bufS[16 + kh + e][16 * j + n];
    }
    b[j] = t;
  }
#pragma unroll
  for (int j = 0; j < 4; ++j) {
    acc[0][j] = wmma_bf16(a0, b[j], acc[0][j]);
    acc[1][j] = wmma_bf16(a1, b[j], acc[1][j]);
  }
}

// One wave computes a 32(M) x 64(N) fp32 tile; the 32(K) x 64(N) B panel is
// shared block-wide: double-buffered in LDS, filled by async global->LDS DMA
// issued cooperatively by all 256 threads, overlapped with WMMA on the other
// stage.  K must be a multiple of 64 (512 or 1024 here) -> loop unrolled x2
// with hard-coded stage buffers.  buf: __shared__ bf16 [2][32][BSTRIDE].
__device__ __forceinline__ void gemm32x64_async(
    const bf16_t* __restrict__ A, int lda,
    const bf16_t* __restrict__ Bg, int ldb, int bcol0,
    int row0, int K, int tid,
    bf16_t (*buf)[32][BSTRIDE], v8f acc[2][4]) {
  int lane = tid & 31;
  int brow = tid >> 3;   // 0..31 : B K-row handled by this thread
  int bseg = tid & 7;    // 0..7  : 16-byte chunk within the 128B row
  const bf16_t* gb = Bg + (size_t)brow * ldb + bcol0 + bseg * 8;
  uint32_t l0 = lds_addr_of(&buf[0][brow][bseg * 8]);
  uint32_t l1 = lds_addr_of(&buf[1][brow][bseg * 8]);

  async_b128_to_lds(l0, gb);                       // stage 0 in flight
  for (int k = 0; k < K; k += 64) {
    // ---- stage 0 holds rows k..k+31 ----
    if (k + 32 < K) {
      async_b128_to_lds(l1, gb + (size_t)(k + 32) * ldb);
      wait_asynccnt_le1();
    } else {
      wait_asynccnt_0();
    }
    __syncthreads();
    stage_compute(A, lda, row0, k, lane, buf[0], acc);
    __syncthreads();
    // ---- stage 1 holds rows k+32..k+63 ----
    if (k + 64 < K) {
      async_b128_to_lds(l0, gb + (size_t)(k + 64) * ldb);
      wait_asynccnt_le1();
    } else {
      wait_asynccnt_0();
    }
    __syncthreads();
    stage_compute(A, lda, row0, k + 32, lane, buf[1], acc);
    __syncthreads();
  }
}

// ---- conversion / transpose kernels ----

__global__ __launch_bounds__(256) void f32_to_bf16_kernel(
    const float* __restrict__ s, bf16_t* __restrict__ d, int n) {
  int i = blockIdx.x * 256 + threadIdx.x;
  if (i < n) d[i] = (bf16_t)s[i];
}

// pos_emb [2047,512] -> bf16 [2048,512], last row zero
__global__ __launch_bounds__(256) void pos_to_bf16_kernel(
    const float* __restrict__ s, bf16_t* __restrict__ d) {
  int i = blockIdx.x * 256 + threadIdx.x;   // 2048*512 threads
  int row = i >> 9;
  d[i] = (row < (2 * TT - 1)) ? (bf16_t)s[i] : (bf16_t)0.0f;
}

// W [512,512] fp32 row-major -> WT bf16 with WT[fin][fout] = W[fout][fin]
__global__ __launch_bounds__(256) void transpose_w_kernel(
    const float* __restrict__ W, bf16_t* __restrict__ WT) {
  int i = blockIdx.x * 256 + threadIdx.x;   // over source
  int r = i >> 9, c = i & 511;
  WT[(size_t)c * FF + r] = (bf16_t)W[i];
}

// ---- projection kernels (grid: (rows/256, 512/64), block 256 = 8 waves) ----

__global__ __launch_bounds__(256) void proj_q_kernel(
    const bf16_t* __restrict__ xh, const bf16_t* __restrict__ WT,
    const float* __restrict__ bq, const float* __restrict__ pbu,
    const float* __restrict__ pbv,
    bf16_t* __restrict__ qu, bf16_t* __restrict__ qv) {
  __shared__ bf16_t bpanel[2][32][BSTRIDE];
  int tid = threadIdx.x, lane = tid & 31, wave = tid >> 5;
  int row0 = blockIdx.x * 256 + wave * 32;
  int col0 = blockIdx.y * 64;
  v8f acc[2][4] = {};
  gemm32x64_async(xh, FF, WT, FF, col0, row0, FF, tid, bpanel, acc);
  int n = lane & 15, mo = (lane >> 4) << 3;
#pragma unroll
  for (int m = 0; m < 2; ++m)
#pragma unroll
    for (int j = 0; j < 4; ++j) {
      int fo = col0 + 16 * j + n;
      int h = fo >> 6, d = fo & 63;
      float bias = bq[fo], bu = pbu[fo], bv = pbv[fo];
#pragma unroll
      for (int r = 0; r < 8; ++r) {
        int rowg = row0 + 16 * m + mo + r;
        int b = rowg >> 10, t = rowg & (TT - 1);
        size_t idx = (((size_t)(b * HH + h) * TT) + t) * DKK + d;
        float qf = acc[m][j][r] + bias;
        qu[idx] = (bf16_t)(qf + bu);
        qv[idx] = (bf16_t)(qf + bv);
      }
    }
}

__global__ __launch_bounds__(256) void proj_k_kernel(
    const bf16_t* __restrict__ xh, const bf16_t* __restrict__ WT,
    const float* __restrict__ bk, bf16_t* __restrict__ kT) {
  __shared__ bf16_t bpanel[2][32][BSTRIDE];
  int tid = threadIdx.x, lane = tid & 31, wave = tid >> 5;
  int row0 = blockIdx.x * 256 + wave * 32;
  int col0 = blockIdx.y * 64;
  v8f acc[2][4] = {};
  gemm32x64_async(xh, FF, WT, FF, col0, row0, FF, tid, bpanel, acc);
  int n = lane & 15, mo = (lane >> 4) << 3;
#pragma unroll
  for (int m = 0; m < 2; ++m)
#pragma unroll
    for (int j = 0; j < 4; ++j) {
      int fo = col0 + 16 * j + n;
      int h = fo >> 6, d = fo & 63;
      float bias = bk[fo];
#pragma unroll
      for (int r = 0; r < 8; ++r) {
        int rowg = row0 + 16 * m + mo + r;
        int b = rowg >> 10, t = rowg & (TT - 1);
        // kT layout [B,H,DK,T] : B-matrix for QK^T (K rows = d, N contiguous)
        kT[(((size_t)(b * HH + h) * DKK) + d) * TT + t] =
            (bf16_t)(acc[m][j][r] + bias);
      }
    }
}

__global__ __launch_bounds__(256) void proj_v_kernel(
    const bf16_t* __restrict__ xh, const bf16_t* __restrict__ WT,
    const float* __restrict__ bv, bf16_t* __restrict__ vout) {
  __shared__ bf16_t bpanel[2][32][BSTRIDE];
  int tid = threadIdx.x, lane = tid & 31, wave = tid >> 5;
  int row0 = blockIdx.x * 256 + wave * 32;
  int col0 = blockIdx.y * 64;
  v8f acc[2][4] = {};
  gemm32x64_async(xh, FF, WT, FF, col0, row0, FF, tid, bpanel, acc);
  int n = lane & 15, mo = (lane >> 4) << 3;
#pragma unroll
  for (int m = 0; m < 2; ++m)
#pragma unroll
    for (int j = 0; j < 4; ++j) {
      int fo = col0 + 16 * j + n;
      int h = fo >> 6, d = fo & 63;
      float bias = bv[fo];
#pragma unroll
      for (int r = 0; r < 8; ++r) {
        int rowg = row0 + 16 * m + mo + r;
        int b = rowg >> 10, t = rowg & (TT - 1);
        // v layout [B,H,T,DK] : B-matrix for attn@V (K rows = s, N contiguous)
        vout[(((size_t)(b * HH + h) * TT) + t) * DKK + d] =
            (bf16_t)(acc[m][j][r] + bias);
      }
    }
}

__global__ __launch_bounds__(256) void proj_p_kernel(
    const bf16_t* __restrict__ ph, const bf16_t* __restrict__ WT,
    bf16_t* __restrict__ pT) {
  __shared__ bf16_t bpanel[2][32][BSTRIDE];
  int tid = threadIdx.x, lane = tid & 31, wave = tid >> 5;
  int row0 = blockIdx.x * 256 + wave * 32;   // p index (padded to 2048)
  int col0 = blockIdx.y * 64;
  v8f acc[2][4] = {};
  gemm32x64_async(ph, FF, WT, FF, col0, row0, FF, tid, bpanel, acc);
  int n = lane & 15, mo = (lane >> 4) << 3;
#pragma unroll
  for (int m = 0; m < 2; ++m)
#pragma unroll
    for (int j = 0; j < 4; ++j) {
      int fo = col0 + 16 * j + n;
      int h = fo >> 6, d = fo & 63;
#pragma unroll
      for (int r = 0; r < 8; ++r) {
        int pidx = row0 + 16 * m + mo + r;
        // pT layout [H,DK,PPAD] : B-matrix for QV.P^T (K rows = d, N contig.)
        pT[(((size_t)h * DKK) + d) * PPAD + pidx] = (bf16_t)acc[m][j][r];
      }
    }
}

// ---- scores: ac + rel-shifted bd, scaled, +mask -> fp32 [B,H,T,T] ----
// grid (T/16/8, T/16, B*H), block 256; wave handles one 16(t)x16(s) tile.
// B tiles here are per-wave unique (k-tile at s0, p-tile at s0-t0) -> direct
// global loads; no block-level sharing to exploit.
__global__ __launch_bounds__(256) void scores_kernel(
    const bf16_t* __restrict__ qu, const bf16_t* __restrict__ qv,
    const bf16_t* __restrict__ kT, const bf16_t* __restrict__ pT,
    const float* __restrict__ mask, float* __restrict__ scores) {
  __shared__ float bdlds[8][16][32];
  int lane = threadIdx.x & 31, wave = threadIdx.x >> 5;
  int bh = blockIdx.z;
  int h = bh & (HH - 1);
  int t0 = blockIdx.y * 16;
  int s0 = (blockIdx.x * 8 + wave) * 16;

  const bf16_t* quB = qu + (size_t)bh * TT * DKK;
  const bf16_t* qvB = qv + (size_t)bh * TT * DKK;
  const bf16_t* kB  = kT + (size_t)bh * DKK * TT;
  const bf16_t* pB  = pT + (size_t)h * DKK * PPAD;

  v16bf au0 = load_a_tile(quB, DKK, t0, 0, lane);
  v16bf au1 = load_a_tile(quB, DKK, t0, 32, lane);
  v16bf av0 = load_a_tile(qvB, DKK, t0, 0, lane);
  v16bf av1 = load_a_tile(qvB, DKK, t0, 32, lane);

  // ac = qu . k   (K = DK = 64 -> two WMMA steps)
  v8f ac = {};
  ac = wmma_bf16(au0, load_b_tile(kB, TT, 0, s0, lane), ac);
  ac = wmma_bf16(au1, load_b_tile(kB, TT, 32, s0, lane), ac);

  // bd strip: p indices pbase..pbase+31 ;  shifted bd[t,s] = bd_raw[t, s-t+T-1]
  int pbase = s0 - t0 + (TT - 16);            // in [0, 2016]; +31 fits PPAD
  v8f bd0 = {}, bd1 = {};
  bd0 = wmma_bf16(av0, load_b_tile(pB, PPAD, 0, pbase, lane), bd0);
  bd0 = wmma_bf16(av1, load_b_tile(pB, PPAD, 32, pbase, lane), bd0);
  bd1 = wmma_bf16(av0, load_b_tile(pB, PPAD, 0, pbase + 16, lane), bd1);
  bd1 = wmma_bf16(av1, load_b_tile(pB, PPAD, 32, pbase + 16, lane), bd1);

  int n = lane & 15, mo = (lane >> 4) << 3;
#pragma unroll
  for (int r = 0; r < 8; ++r) {
    bdlds[wave][mo + r][n]      = bd0[r];
    bdlds[wave][mo + r][16 + n] = bd1[r];
  }
  __syncthreads();

  const float scale = 0.125f;                 // 1/sqrt(DK)
#pragma unroll
  for (int r = 0; r < 8; ++r) {
    int m = mo + r;
    float bd = bdlds[wave][m][n - m + 15];    // diagonal gather (rel_shift)
    float val = (ac[r] + bd) * scale + mask[(size_t)(t0 + m) * TT + s0 + n];
    scores[((size_t)bh * TT + t0 + m) * TT + s0 + n] = val;
  }
}

// ---- softmax: one wave per 1024-wide row; fp32 in, bf16 out ----
__global__ __launch_bounds__(256) void softmax_kernel(
    const float* __restrict__ scores, bf16_t* __restrict__ attn) {
  int lane = threadIdx.x & 31, wave = threadIdx.x >> 5;
  size_t row = (size_t)blockIdx.x * 8 + wave;   // B*H*T rows
  const float* srow = scores + row * TT;
  float vals[32];
  float mx = -1e30f;
#pragma unroll
  for (int i = 0; i < 32; ++i) {
    vals[i] = srow[lane + 32 * i];
    mx = fmaxf(mx, vals[i]);
  }
#pragma unroll
  for (int off = 16; off > 0; off >>= 1) mx = fmaxf(mx, __shfl_xor(mx, off, 32));
  float sum = 0.0f;
#pragma unroll
  for (int i = 0; i < 32; ++i) { vals[i] = __expf(vals[i] - mx); sum += vals[i]; }
#pragma unroll
  for (int off = 16; off > 0; off >>= 1) sum += __shfl_xor(sum, off, 32);
  float inv = 1.0f / sum;
  bf16_t* arow = attn + row * TT;
#pragma unroll
  for (int i = 0; i < 32; ++i) arow[lane + 32 * i] = (bf16_t)(vals[i] * inv);
}

// ---- ctx = attn @ v : grid (T/256, 1, B*H), wave -> 32(t) x 64(DK) ----
// V panel (32 x 64) is identical across the 8 waves -> async LDS staging.
__global__ __launch_bounds__(256) void av_kernel(
    const bf16_t* __restrict__ attn, const bf16_t* __restrict__ v,
    bf16_t* __restrict__ ctx) {
  __shared__ bf16_t bpanel[2][32][BSTRIDE];
  int tid = threadIdx.x, lane = tid & 31, wave = tid >> 5;
  int bh = blockIdx.z;
  int b = bh >> 3, h = bh & (HH - 1);
  int t0 = blockIdx.x * 256 + wave * 32;
  const bf16_t* Abh = attn + (size_t)bh * TT * TT;
  const bf16_t* Vbh = v + (size_t)bh * TT * DKK;
  v8f acc[2][4] = {};
  gemm32x64_async(Abh, TT, Vbh, DKK, 0, t0, TT, tid, bpanel, acc);
  int n = lane & 15, mo = (lane >> 4) << 3;
#pragma unroll
  for (int m = 0; m < 2; ++m)
#pragma unroll
    for (int j = 0; j < 4; ++j)
#pragma unroll
      for (int r = 0; r < 8; ++r)
        ctx[((size_t)b * TT + t0 + 16 * m + mo + r) * FF + h * DKK + 16 * j + n] =
            (bf16_t)acc[m][j][r];
}

// ---- final projection: out = ctx @ Wout^T + bout (fp32 out) ----
__global__ __launch_bounds__(256) void proj_out_kernel(
    const bf16_t* __restrict__ ctx, const bf16_t* __restrict__ WT,
    const float* __restrict__ bout, float* __restrict__ out) {
  __shared__ bf16_t bpanel[2][32][BSTRIDE];
  int tid = threadIdx.x, lane = tid & 31, wave = tid >> 5;
  int row0 = blockIdx.x * 256 + wave * 32;
  int col0 = blockIdx.y * 64;
  v8f acc[2][4] = {};
  gemm32x64_async(ctx, FF, WT, FF, col0, row0, FF, tid, bpanel, acc);
  int n = lane & 15, mo = (lane >> 4) << 3;
#pragma unroll
  for (int m = 0; m < 2; ++m)
#pragma unroll
    for (int j = 0; j < 4; ++j) {
      int fo = col0 + 16 * j + n;
      float bias = bout[fo];
#pragma unroll
      for (int r = 0; r < 8; ++r)
        out[(size_t)(row0 + 16 * m + mo + r) * FF + fo] = acc[m][j][r] + bias;
    }
}

// ---- host-side orchestration ----

extern "C" void kernel_launch(void* const* d_in, const int* in_sizes, int n_in,
                              void* d_out, int out_size, void* d_ws, size_t ws_size,
                              hipStream_t stream) {
  const float* x    = (const float*)d_in[0];
  const float* pos  = (const float*)d_in[1];
  const float* mask = (const float*)d_in[2];
  const float* Wq   = (const float*)d_in[3];
  const float* bq   = (const float*)d_in[4];
  const float* Wk   = (const float*)d_in[5];
  const float* bk   = (const float*)d_in[6];
  const float* Wv   = (const float*)d_in[7];
  const float* bv   = (const float*)d_in[8];
  const float* Wpos = (const float*)d_in[9];
  const float* Wout = (const float*)d_in[10];
  const float* bout = (const float*)d_in[11];
  const float* pbu  = (const float*)d_in[12];
  const float* pbv  = (const float*)d_in[13];

  char* ws = (char*)d_ws;
  auto alloc = [&](size_t bytes) {
    char* p = ws;
    ws += (bytes + 255) & ~(size_t)255;
    return p;
  };

  const size_t NROW = (size_t)BB * TT;            // 4096
  bf16_t* xh    = (bf16_t*)alloc(NROW * FF * 2);
  bf16_t* ph    = (bf16_t*)alloc((size_t)PPAD * FF * 2);
  bf16_t* WqT   = (bf16_t*)alloc((size_t)FF * FF * 2);
  bf16_t* WkT   = (bf16_t*)alloc((size_t)FF * FF * 2);
  bf16_t* WvT   = (bf16_t*)alloc((size_t)FF * FF * 2);
  bf16_t* WpT   = (bf16_t*)alloc((size_t)FF * FF * 2);
  bf16_t* WoT   = (bf16_t*)alloc((size_t)FF * FF * 2);
  bf16_t* qu    = (bf16_t*)alloc((size_t)BB * HH * TT * DKK * 2);
  bf16_t* qv    = (bf16_t*)alloc((size_t)BB * HH * TT * DKK * 2);
  bf16_t* kT    = (bf16_t*)alloc((size_t)BB * HH * DKK * TT * 2);
  bf16_t* vv    = (bf16_t*)alloc((size_t)BB * HH * TT * DKK * 2);
  bf16_t* pT    = (bf16_t*)alloc((size_t)HH * DKK * PPAD * 2);
  float*  scr   = (float*) alloc((size_t)BB * HH * TT * TT * 4);
  bf16_t* attn  = (bf16_t*)alloc((size_t)BB * HH * TT * TT * 2);
  bf16_t* ctx   = (bf16_t*)alloc(NROW * FF * 2);

  // 1) precision conversion / weight transpose
  f32_to_bf16_kernel<<<(NROW * FF + 255) / 256, 256, 0, stream>>>(x, xh, NROW * FF);
  pos_to_bf16_kernel<<<((size_t)PPAD * FF) / 256, 256, 0, stream>>>(pos, ph);
  transpose_w_kernel<<<(FF * FF) / 256, 256, 0, stream>>>(Wq, WqT);
  transpose_w_kernel<<<(FF * FF) / 256, 256, 0, stream>>>(Wk, WkT);
  transpose_w_kernel<<<(FF * FF) / 256, 256, 0, stream>>>(Wv, WvT);
  transpose_w_kernel<<<(FF * FF) / 256, 256, 0, stream>>>(Wpos, WpT);
  transpose_w_kernel<<<(FF * FF) / 256, 256, 0, stream>>>(Wout, WoT);

  // 2) projections (WMMA bf16, async-LDS staged weight panels)
  dim3 gproj(NROW / 256, FF / 64);
  proj_q_kernel<<<gproj, 256, 0, stream>>>(xh, WqT, bq, pbu, pbv, qu, qv);
  proj_k_kernel<<<gproj, 256, 0, stream>>>(xh, WkT, bk, kT);
  proj_v_kernel<<<gproj, 256, 0, stream>>>(xh, WvT, bv, vv);
  dim3 gpos(PPAD / 256, FF / 64);
  proj_p_kernel<<<gpos, 256, 0, stream>>>(ph, WpT, pT);

  // 3) scores with rel-shift (WMMA), fp32 into L2-resident scratch
  dim3 gsc(TT / 16 / 8, TT / 16, BB * HH);
  scores_kernel<<<gsc, 256, 0, stream>>>(qu, qv, kT, pT, mask, scr);

  // 4) softmax -> bf16 attn
  softmax_kernel<<<(BB * HH * TT) / 8, 256, 0, stream>>>(scr, attn);

  // 5) ctx = attn @ v (WMMA, async-LDS staged V panels)
  dim3 gav(TT / 256, 1, BB * HH);
  av_kernel<<<gav, 256, 0, stream>>>(attn, vv, ctx);

  // 6) output projection (WMMA), fp32 result
  proj_out_kernel<<<gproj, 256, 0, stream>>>(ctx, WoT, bout, (float*)d_out);
}
